// GCNEncoder_80255758893845
// MI455X (gfx1250) — matrix operand order repaired and checked by reading, not compiled, not run on hardware
//
#include <hip/hip_runtime.h>

// ---------------------------------------------------------------------------
// GCN encoder for MI455X (gfx1250, wave32).
//   h  = ReLU( Ahat @ (x @ W1) + b1 )
//   o  = Ahat @ (h @ W2) + b2 ;  out = o / max(||o||2, 1e-12)
// Ahat = sym-normalized adjacency with self loops (deg at dst).
// GEMMs run on v_wmma_f32_16x16x32_bf16; aggregation via L2 f32 atomics.
// ---------------------------------------------------------------------------

typedef __attribute__((ext_vector_type(16))) __bf16 v16bf;
typedef __attribute__((ext_vector_type(8)))  float  v8f;

union BF16Frag { uint4 u[2]; v16bf v; };

__device__ __forceinline__ __bf16 f2bf(float f) {
  union { float f; unsigned u; } a; a.f = f;
  unsigned r = a.u + 0x7FFFu + ((a.u >> 16) & 1u);   // round-to-nearest-even
  unsigned short h = (unsigned short)(r >> 16);
  return __builtin_bit_cast(__bf16, h);
}

#define NOUT 256   // output width of both layers (HID == OUT == 256)

// ---------------- degree / normalization ----------------------------------
__global__ __launch_bounds__(256) void k_deg_init(float* deg, int n) {
  int i = blockIdx.x * 256 + threadIdx.x;
  if (i < n) deg[i] = 1.0f;                       // self loop contributes 1
}

__global__ __launch_bounds__(256) void k_deg_edges(const long long* __restrict__ ei,
                                                   float* deg, int E) {
  int e = blockIdx.x * 256 + threadIdx.x;
  if (e < E) {
    long long d = ei[(size_t)E + e];              // dst row of edge_index
    atomicAdd(&deg[d], 1.0f);
  }
}

__global__ __launch_bounds__(256) void k_dinv(float* deg, int n) {
  int i = blockIdx.x * 256 + threadIdx.x;
  if (i < n) {
    float d = deg[i];
    deg[i] = (d > 0.0f) ? rsqrtf(d) : 0.0f;       // in-place: deg -> dinv
  }
}

// ---------------- conversions ---------------------------------------------
__global__ __launch_bounds__(256) void k_cvt_f32_bf16(const float* __restrict__ src,
                                                      __bf16* __restrict__ dst, int n) {
  int i = blockIdx.x * 256 + threadIdx.x;
  if (i < n) dst[i] = f2bf(src[i]);
}

// W [K, 256] row-major  ->  Wt [256, K] bf16 (so B frags are K-contiguous)
__global__ __launch_bounds__(256) void k_cvt_w_t(const float* __restrict__ W,
                                                 __bf16* __restrict__ Wt, int K) {
  int idx = blockIdx.x * 256 + threadIdx.x;       // idx = n*K + k
  if (idx < K * NOUT) {
    int n = idx / K, k = idx - n * K;
    Wt[idx] = f2bf(W[(size_t)k * NOUT + n]);
  }
}

// ---------------- WMMA GEMM:  C[M,256] = A[M,K](bf16) @ W[K,256] -----------
// Block: 256 threads (8 waves). Block tile 128x256; wave tile 16x256.
#define KSTEP 32
#define SW_STRIDE 40   // KSTEP + 8 bf16 pad; keeps 16B alignment, staggers banks

__global__ __launch_bounds__(256) void k_gemm_bf16(const __bf16* __restrict__ A,
                                                   const __bf16* __restrict__ Wt,
                                                   float* __restrict__ C,
                                                   int M, int K) {
  __shared__ __bf16 sW[NOUT][SW_STRIDE];          // 20 KB
  const int tid    = threadIdx.x;
  const int lane   = tid & 31;
  const int wave   = tid >> 5;                    // 0..7
  const int laneLo = lane & 15;
  const bool hiHalf = (lane >= 16);
  const int rowBase = blockIdx.x * 128 + wave * 16;
  const int m = rowBase + laneLo;                 // row this lane loads
  const bool mOK = (m < M);

  v8f acc[16] = {};                               // 16 n-tiles of 16x16 f32

  for (int k0 = 0; k0 < K; k0 += KSTEP) {
    // Stage Wt chunk: thread t copies Wt[t][k0..k0+31] (64 B)
    {
      const uint4* s = (const uint4*)(Wt + (size_t)tid * K + k0);
      uint4* d = (uint4*)(&sW[tid][0]);
      d[0] = s[0]; d[1] = s[1]; d[2] = s[2]; d[3] = s[3];
    }
    __syncthreads();

    // A fragment per CDNA5 16-bit A layout (16x32, row m, K window k0..k0+31)
    BF16Frag af;
    if (mOK) {
      const __bf16* ar = A + (size_t)m * K + k0 + (hiHalf ? 8 : 0);
      af.u[0] = *(const uint4*)(ar);        // K +0..7   (hi: +8..15)
      af.u[1] = *(const uint4*)(ar + 16);   // K +16..23 (hi: +24..31)
    } else {
      af.u[0] = make_uint4(0u, 0u, 0u, 0u);
      af.u[1] = make_uint4(0u, 0u, 0u, 0u);
    }

    #pragma unroll
    for (int t = 0; t < 16; ++t) {
      const int n = t * 16 + laneLo;        // this lane's B column
      // B frag: lanes 0-15 hold K 0..7,16..23; lanes 16-31 hold K 8..15,24..31
      const uint4* bs = (const uint4*)(&sW[n][hiHalf ? 8 : 0]);
      BF16Frag bf;
      bf.u[0] = bs[0];
      bf.u[1] = bs[2];
      acc[t] = __builtin_amdgcn_wmma_f32_16x16x32_bf16(
          false, af.v, false, bf.v, (short)0, acc[t], false, false);
    }
    __syncthreads();
  }

  // Store: C-layout VGPR j -> row j (lanes 0-15) / row j+8 (lanes 16-31)
  #pragma unroll
  for (int t = 0; t < 16; ++t) {
    const int n = t * 16 + laneLo;
    #pragma unroll
    for (int j = 0; j < 8; ++j) {
      int row = rowBase + j + (hiHalf ? 8 : 0);
      if (row < M) C[(size_t)row * NOUT + n] = acc[t][j];
    }
  }
}

// ---------------- aggregation ----------------------------------------------
// agg[n,:] = dinv[n]^2 * t[n,:]   (self-loop term; also zero-inits agg)
__global__ __launch_bounds__(256) void k_agg_init(const float* __restrict__ t,
                                                  const float* __restrict__ dinv,
                                                  float* __restrict__ agg, int N) {
  int idx = blockIdx.x * 256 + threadIdx.x;       // N*64 float4 slots
  if (idx < N * 64) {
    int n = idx >> 6;
    int c = (idx & 63) * 4;
    float w = dinv[n] * dinv[n];
    float4 v = *(const float4*)(t + (size_t)n * NOUT + c);
    float4 o; o.x = w * v.x; o.y = w * v.y; o.z = w * v.z; o.w = w * v.w;
    *(float4*)(agg + (size_t)n * NOUT + c) = o;
  }
}

// Per edge (64 lanes/edge): agg[dst,:] += dinv[src]*dinv[dst] * t[src,:]
__global__ __launch_bounds__(256) void k_scatter(const long long* __restrict__ ei,
                                                 const float* __restrict__ dinv,
                                                 const float* __restrict__ t,
                                                 float* __restrict__ agg, int E) {
  int e = blockIdx.x * 4 + (threadIdx.x >> 6);
  if (e >= E) return;
  int c = (threadIdx.x & 63) * 4;
  long long s = ei[e];
  long long d = ei[(size_t)E + e];
  float w = dinv[s] * dinv[d];
  float4 v = *(const float4*)(t + (size_t)s * NOUT + c);   // L2-resident gather
  float* o = agg + (size_t)d * NOUT + c;
  atomicAdd(o + 0, w * v.x);                               // global_atomic_add_f32
  atomicAdd(o + 1, w * v.y);
  atomicAdd(o + 2, w * v.z);
  atomicAdd(o + 3, w * v.w);
}

// h_bf16 = bf16( relu(agg + b1) )  -> input of GEMM2
__global__ __launch_bounds__(256) void k_relu_bias_cvt(const float* __restrict__ agg,
                                                       const float* __restrict__ b,
                                                       __bf16* __restrict__ h, int total) {
  int i = blockIdx.x * 256 + threadIdx.x;
  if (i < total) {
    float v = agg[i] + b[i & (NOUT - 1)];
    h[i] = f2bf(v > 0.0f ? v : 0.0f);
  }
}

// out[n,:] = (agg[n,:]+b2) / max(||agg[n,:]+b2||2, 1e-12);  one block per node
__global__ __launch_bounds__(256) void k_norm(const float* __restrict__ agg,
                                              const float* __restrict__ b,
                                              float* __restrict__ out, int N) {
  __shared__ float red[256];
  const int n = blockIdx.x, c = threadIdx.x;
  float v = agg[(size_t)n * NOUT + c] + b[c];
  red[c] = v * v;
  __syncthreads();
  #pragma unroll
  for (int s = 128; s > 0; s >>= 1) {
    if (c < s) red[c] += red[c + s];
    __syncthreads();
  }
  float nrm = sqrtf(red[0]);
  out[(size_t)n * NOUT + c] = v / fmaxf(nrm, 1e-12f);
}

// ---------------------------------------------------------------------------
extern "C" void kernel_launch(void* const* d_in, const int* in_sizes, int n_in,
                              void* d_out, int out_size, void* d_ws, size_t ws_size,
                              hipStream_t stream) {
  const float*     x   = (const float*)d_in[0];
  const long long* ei  = (const long long*)d_in[1];
  const float*     W1  = (const float*)d_in[2];
  const float*     b1  = (const float*)d_in[3];
  const float*     W2  = (const float*)d_in[4];
  const float*     b2  = (const float*)d_in[5];
  float*           out = (float*)d_out;

  const int IN_DIM = 128;
  const int N = in_sizes[0] / IN_DIM;
  const int E = in_sizes[1] / 2;

  // Workspace carve-up (256B aligned)
  auto align = [](size_t v) { return (v + 255) & ~(size_t)255; };
  char* p = (char*)d_ws;
  size_t off = 0;
  float*  dinv = (float*)(p + off);  off = align(off + (size_t)N * 4);
  __bf16* xb   = (__bf16*)(p + off); off = align(off + (size_t)N * IN_DIM * 2);
  __bf16* w1t  = (__bf16*)(p + off); off = align(off + (size_t)NOUT * IN_DIM * 2);
  __bf16* w2t  = (__bf16*)(p + off); off = align(off + (size_t)NOUT * NOUT * 2);
  float*  tbuf = (float*)(p + off);  off = align(off + (size_t)N * NOUT * 4);
  float*  agg  = (float*)(p + off);  off = align(off + (size_t)N * NOUT * 4);
  __bf16* hb   = (__bf16*)(p + off); off = align(off + (size_t)N * NOUT * 2);
  (void)ws_size; (void)n_in; (void)out_size;

  const int TPB = 256;
  auto cdiv = [](int a, int b) { return (a + b - 1) / b; };

  // --- normalization coefficients (shared by both layers) ---
  k_deg_init <<<cdiv(N, TPB), TPB, 0, stream>>>(dinv, N);
  k_deg_edges<<<cdiv(E, TPB), TPB, 0, stream>>>(ei, dinv, E);
  k_dinv     <<<cdiv(N, TPB), TPB, 0, stream>>>(dinv, N);

  // --- bf16 conversions ---
  k_cvt_f32_bf16<<<cdiv(N * IN_DIM, TPB), TPB, 0, stream>>>(x, xb, N * IN_DIM);
  k_cvt_w_t<<<cdiv(IN_DIM * NOUT, TPB), TPB, 0, stream>>>(W1, w1t, IN_DIM);
  k_cvt_w_t<<<cdiv(NOUT * NOUT, TPB), TPB, 0, stream>>>(W2, w2t, NOUT);

  // --- layer 1 ---
  k_gemm_bf16<<<cdiv(N, 128), TPB, 0, stream>>>(xb, w1t, tbuf, N, IN_DIM);
  k_agg_init <<<cdiv(N * 64, TPB), TPB, 0, stream>>>(tbuf, dinv, agg, N);
  k_scatter  <<<cdiv(E, 4), TPB, 0, stream>>>(ei, dinv, tbuf, agg, E);
  k_relu_bias_cvt<<<cdiv(N * NOUT, TPB), TPB, 0, stream>>>(agg, b1, hb, N * NOUT);

  // --- layer 2 ---
  k_gemm_bf16<<<cdiv(N, 128), TPB, 0, stream>>>(hb, w2t, tbuf, N, NOUT);
  k_agg_init <<<cdiv(N * 64, TPB), TPB, 0, stream>>>(tbuf, dinv, agg, N);
  k_scatter  <<<cdiv(E, 4), TPB, 0, stream>>>(ei, dinv, tbuf, agg, E);

  // --- bias + L2 normalize ---
  k_norm<<<N, TPB, 0, stream>>>(agg, b2, out, N);
}
